// SNN_2817498546577
// MI455X (gfx1250) — matrix operand order repaired and checked by reading, not compiled, not run on hardware
//
#include <hip/hip_runtime.h>
#include <hip/hip_bf16.h>

// ---------------------------------------------------------------------------
// SNN: 3-layer synaptic LIF network, T=100, B=16, NIN=NH=10000, NOUT=5.
// Layer-1 spikes depend only on x -> precompute all T of them and collapse
// the 100 per-step [16x10000]x[10000x10000] matmuls into ONE
// [1664x10016]x[10016x10000] bf16 WMMA GEMM (W1 read once from HBM, spikes
// exact in bf16). Same trick for layer-2; only the 80-state layer-3 scan is
// sequential. GEMM: M padded to 104 tiles (8 waves x 13 tiles), A fragments
// rotate through 4 register buffers at prefetch distance 3 with a phase that
// tracks chunk index (13 = 1 mod 4), B fragment ping-pongs between two
// registers via 4x-unrolled chunk loop -> no copies, no full-drain waits.
// ---------------------------------------------------------------------------

#define T_STEPS 100
#define BSZ     16
#define NIN     10000
#define NH      10000
#define NOUT    5
#define LDK     10016          // K padded to a multiple of 32 (WMMA K)
#define MROWS   1600           // T*B
#define MPAD    1664           // 104 tiles = 8 waves * 13 tiles (zero-padded)
#define KCH     (LDK / 32)     // 313 k-chunks; KCH-1 = 312 = 4*78
#define MT      13             // M tiles per wave

typedef __attribute__((ext_vector_type(16))) __bf16 v16bf;
typedef __attribute__((ext_vector_type(8)))  __bf16 v8bf;
typedef __attribute__((ext_vector_type(4)))  __bf16 v4bf;
typedef __attribute__((ext_vector_type(8)))  float  v8f;
typedef __attribute__((ext_vector_type(4)))  float  v4f;

union AFrag { v16bf v; v8bf h[2]; };

// ---------------------------------------------------------------------------
// Layer-1 temporal scan (parallel over B x LDK; pad columns write zeros).
// spk1 stored row-major [t*B+b][LDK] as bf16 {0,1} -> exact.
// ---------------------------------------------------------------------------
__global__ __launch_bounds__(256) void snn_layer1(const float* __restrict__ x,
                                                  __bf16* __restrict__ spk1) {
  const int idx = blockIdx.x * 256 + threadIdx.x;   // < BSZ*LDK (exact grid)
  const int b = idx / LDK;
  const int i = idx % LDK;
  const __bf16 one = (__bf16)1.0f;
  const __bf16 zer = (__bf16)0.0f;
  __bf16* sp = spk1 + (size_t)b * LDK + i;
  if (i < NIN) {
    const float* xp = x + (size_t)b * NIN + i;
    float s = 0.f, m = 0.f;
    for (int t = 0; t < T_STEPS; ++t) {
      const float inp   = xp[(size_t)t * (BSZ * NIN)];
      const float reset = (m > 1.0f) ? 1.0f : 0.0f;   // uses mem BEFORE update
      s = 0.9f * s + inp;
      m = 0.9f * m + s - reset;                        // THR == 1
      sp[(size_t)t * (BSZ * LDK)] = (m > 1.0f) ? one : zer;
    }
  } else {
    for (int t = 0; t < T_STEPS; ++t) sp[(size_t)t * (BSZ * LDK)] = zer;
  }
}

// ---------------------------------------------------------------------------
// Zero the M-padding rows [MROWS, MPAD) of spk1 (64 rows x LDK, v8bf stores).
// ---------------------------------------------------------------------------
__global__ __launch_bounds__(256) void snn_padzero(__bf16* __restrict__ spk1) {
  const int idx = blockIdx.x * 256 + threadIdx.x;   // < (MPAD-MROWS)*LDK/8
  const v8bf z = {(__bf16)0.f, (__bf16)0.f, (__bf16)0.f, (__bf16)0.f,
                  (__bf16)0.f, (__bf16)0.f, (__bf16)0.f, (__bf16)0.f};
  *(v8bf*)(spk1 + (size_t)MROWS * LDK + (size_t)idx * 8) = z;
}

// ---------------------------------------------------------------------------
// W1 fp32 [NH][NIN] -> bf16 [NH][LDK] (same layout, K-padded with zeros).
// Kept N-major so WMMA B-fragments are K-contiguous per lane (no transpose).
// ---------------------------------------------------------------------------
__global__ __launch_bounds__(256) void snn_w1bf16(const float* __restrict__ W1,
                                                  __bf16* __restrict__ W1b) {
  const size_t idx = (size_t)blockIdx.x * 256 + threadIdx.x;
  const int n = (int)(idx / (LDK / 4));
  if (n >= NH) return;
  const int k = ((int)(idx % (LDK / 4))) << 2;
  v4bf o;
  if (k < NIN) {                               // chunks never straddle NIN (4 | 10000)
    const v4f v = *(const v4f*)(W1 + (size_t)n * NIN + k);
    o[0] = (__bf16)v[0]; o[1] = (__bf16)v[1];
    o[2] = (__bf16)v[2]; o[3] = (__bf16)v[3];
  } else {
    o[0] = o[1] = o[2] = o[3] = (__bf16)0.0f;
  }
  *(v4bf*)(W1b + (size_t)n * LDK + k) = o;
}

// ---------------------------------------------------------------------------
// Big GEMM: C[1600][NH] = A[MPAD][LDK] * B^T, B stored [NH][LDK].
// 625 blocks (one 16-wide N tile each); 8 waves x 13 M-tiles each (exact).
// B is read once from HBM; A (33 MB) stays L2-resident.
// ---------------------------------------------------------------------------

// Inline A-fragment load: row M = 16*mt+lm; halves 0..7 at K=kbase+16*lh,
// halves 8..15 at K=kbase+16*lh+8 (contiguous K, matching B fragment).
#define LOADA(dst, ti, kbase)                                                  \
  do {                                                                         \
    const __bf16* _ap = A +                                                    \
        (size_t)(((wv + ((ti) << 3)) << 4) + lm) * LDK + (kbase) + (lh << 4);  \
    (dst).h[0] = *(const v8bf*)(_ap);                                          \
    (dst).h[1] = *(const v8bf*)(_ap + 8);                                      \
  } while (0)

// One k-chunk: 13 WMMAs against BCUR; A-fragment buffer index for tile j of
// chunk kc is (j + kc) & 3 (phase-consistent since 13 = 1 mod 4). Loads run
// at prefetch distance 3, wrapping into tiles 0..2 of the next chunk.
// If LOADB, the next chunk's B fragment is loaded (into the other register)
// at the TOP of this chunk, a full chunk ahead of its first consumer.
template <int PH, bool LOADB, bool WRAPA>
__device__ __forceinline__ void gemm_chunk(const __bf16* __restrict__ A,
                                           const __bf16*& bp, const int kb,
                                           const int wv, const int lm, const int lh,
                                           const v16bf& bcur, v16bf& bnxt,
                                           AFrag (&abuf)[4], v8f (&acc)[MT]) {
  if (LOADB) {
    bnxt = *(const v16bf*)(bp);
    bp += 32;
  }
  __builtin_prefetch(bp + 96, 0, 1);                 // stream far-ahead B
#pragma unroll
  for (int i = 0; i < MT; ++i) {
    const int pi = i + 3;                            // prefetch distance 3
    if (pi < MT) {
      LOADA(abuf[(pi + PH) & 3], pi, kb);            // same chunk
    } else if (WRAPA) {
      LOADA(abuf[(pi + PH) & 3], pi - MT, kb + 32);  // tiles 0..2 of next chunk
    }
    acc[i] = __builtin_amdgcn_wmma_f32_16x16x32_bf16(
        false, abuf[(i + PH) & 3].v, false, bcur, (short)0, acc[i], false, false);
  }
}

__global__ __launch_bounds__(256) void snn_gemm1(const __bf16* __restrict__ A,
                                                 const __bf16* __restrict__ Bm,
                                                 float* __restrict__ C) {
  const int nb   = blockIdx.x << 4;                  // N tile base, 0..624*16
  const int lane = threadIdx.x & 31;
  const int lm   = lane & 15;
  const int lh   = lane >> 4;
  // wave id as a scalar (uniform) - used only for tile numbering
  const int wv = __builtin_amdgcn_readfirstlane((int)(threadIdx.x >> 5)); // 0..7

  const v8f zero8 = {0.f, 0.f, 0.f, 0.f, 0.f, 0.f, 0.f, 0.f};
  v8f acc[MT];
#pragma unroll
  for (int i = 0; i < MT; ++i) acc[i] = zero8;

  // B fragment: lane holds N = nb+lm, K = kb + 16*lh .. +15 (contiguous 32B)
  const __bf16* bp = Bm + (size_t)(nb + lm) * LDK + (lh << 4);

  // ---- pipeline prologue: B of chunk 0, A tiles 0..2 of chunk 0 ----
  v16bf b0 = *(const v16bf*)(bp);
  bp += 32;
  v16bf b1 = b0;                                     // defined before first use
  AFrag abuf[4];
  LOADA(abuf[0], 0, 0);
  LOADA(abuf[1], 1, 0);
  LOADA(abuf[2], 2, 0);

  // ---- steady state: 312 chunks, 4x unrolled (phase + B ping-pong) ----
  for (int kc = 0; kc < KCH - 1; kc += 4) {
    const int kb = kc << 5;
    gemm_chunk<0, true, true>(A, bp, kb,      wv, lm, lh, b0, b1, abuf, acc);
    gemm_chunk<1, true, true>(A, bp, kb + 32, wv, lm, lh, b1, b0, abuf, acc);
    gemm_chunk<2, true, true>(A, bp, kb + 64, wv, lm, lh, b0, b1, abuf, acc);
    gemm_chunk<3, true, true>(A, bp, kb + 96, wv, lm, lh, b1, b0, abuf, acc);
  }
  // ---- epilogue chunk 312: phase 312&3==0, even parity -> b0, no loads ----
  gemm_chunk<0, false, false>(A, bp, (KCH - 1) << 5, wv, lm, lh, b0, b1, abuf, acc);

#pragma unroll
  for (int i = 0; i < MT; ++i) {
    const int mt = wv + (i << 3);
    if (mt < 100) {                                  // skip M-pad tiles (store only)
      const int n = nb + lm;
#pragma unroll
      for (int r = 0; r < 8; ++r) {                  // D: M = r + 8*lh, N = lm
        const int m = (mt << 4) + r + (lh << 3);
        C[(size_t)m * NH + n] = acc[i][r];
      }
    }
  }
}

// ---------------------------------------------------------------------------
// Layer-2 temporal scan (parallel over B x NH), cur2 is fully precomputed.
// ---------------------------------------------------------------------------
__global__ __launch_bounds__(256) void snn_layer2(const float* __restrict__ cur2,
                                                  __bf16* __restrict__ spk2) {
  const int idx = blockIdx.x * 256 + threadIdx.x;    // < BSZ*NH (exact grid)
  const int b = idx / NH;
  const int j = idx % NH;
  const __bf16 one = (__bf16)1.0f;
  const __bf16 zer = (__bf16)0.0f;
  const float* cp = cur2 + (size_t)b * NH + j;
  __bf16* sp = spk2 + (size_t)b * NH + j;
  float s = 0.f, m = 0.f;
  for (int t = 0; t < T_STEPS; ++t) {
    const float inp   = cp[(size_t)t * (BSZ * NH)];
    const float reset = (m > 1.0f) ? 1.0f : 0.0f;
    s = 0.9f * s + inp;
    m = 0.9f * m + s - reset;
    sp[(size_t)t * (BSZ * NH)] = (m > 1.0f) ? one : zer;
  }
}

// ---------------------------------------------------------------------------
// cur3[r][o] = sum_j spk2[r][j] * W2[o][j]  (N=5 -> VALU reduction per row)
// ---------------------------------------------------------------------------
__global__ __launch_bounds__(256) void snn_cur3(const __bf16* __restrict__ spk2,
                                                const float* __restrict__ W2,
                                                float* __restrict__ cur3) {
  const int r   = blockIdx.x;                        // 0..1599
  const int tid = threadIdx.x;
  float p[NOUT] = {0.f, 0.f, 0.f, 0.f, 0.f};
  const __bf16* row = spk2 + (size_t)r * NH;
  for (int j = tid; j < NH; j += 256) {
    const float sv = (float)row[j];
#pragma unroll
    for (int o = 0; o < NOUT; ++o) p[o] += sv * W2[(size_t)o * NH + j];
  }
  __shared__ float red[NOUT * 256];
#pragma unroll
  for (int o = 0; o < NOUT; ++o) red[o * 256 + tid] = p[o];
  __syncthreads();
  for (int st = 128; st > 0; st >>= 1) {
    if (tid < st) {
#pragma unroll
      for (int o = 0; o < NOUT; ++o)
        red[o * 256 + tid] += red[o * 256 + tid + st];
    }
    __syncthreads();
  }
  if (tid < NOUT) cur3[(size_t)r * NOUT + tid] = red[tid * 256];
}

// ---------------------------------------------------------------------------
// Layer-3 scan (truly sequential, 80 states) + spike-count normalization.
// ---------------------------------------------------------------------------
__global__ __launch_bounds__(128) void snn_layer3(const float* __restrict__ cur3,
                                                  float* __restrict__ out) {
  __shared__ float accs[BSZ][NOUT];
  __shared__ float rs[BSZ];
  const int tid = threadIdx.x;
  if (tid < BSZ * NOUT) {
    const int b = tid / NOUT;
    const int o = tid % NOUT;
    float s = 0.f, m = 0.f, acc = 0.f;
    for (int t = 0; t < T_STEPS; ++t) {
      const float inp   = cur3[(size_t)(t * BSZ + b) * NOUT + o];
      const float reset = (m > 1.0f) ? 1.0f : 0.0f;
      s = 0.9f * s + inp;
      m = 0.9f * m + s - reset;
      acc += (m > 1.0f) ? 1.0f : 0.0f;
    }
    accs[b][o] = acc;
  }
  __syncthreads();
  if (tid < BSZ) {
    float sum = 0.f;
#pragma unroll
    for (int o = 0; o < NOUT; ++o) sum += accs[tid][o];
    rs[tid] = (sum == 0.f) ? 1.0f : sum;
  }
  __syncthreads();
  if (tid < BSZ * NOUT) {
    const int b = tid / NOUT;
    const int o = tid % NOUT;
    out[b * NOUT + o] = accs[b][o] / rs[b];
  }
}

// ---------------------------------------------------------------------------
// Launch
// ---------------------------------------------------------------------------
extern "C" void kernel_launch(void* const* d_in, const int* in_sizes, int n_in,
                              void* d_out, int out_size, void* d_ws, size_t ws_size,
                              hipStream_t stream) {
  const float* x  = (const float*)d_in[0];   // [T,B,NIN]
  const float* W1 = (const float*)d_in[1];   // [NH,NIN]
  const float* W2 = (const float*)d_in[2];   // [NOUT,NH]
  float* out = (float*)d_out;                // [B,NOUT]
  char* ws = (char*)d_ws;

  // 256B-aligned workspace carve-up (total ~329.7 MB)
  __bf16* SPK1 = (__bf16*)(ws + 0);              //  1664*10016 bf16 = 33,333,248 B
  __bf16* W1b  = (__bf16*)(ws + 33333248);       // 10000*10016 bf16 = 200,320,000 B
  float*  CUR2 = (float*) (ws + 233653248);      //  1600*10000 f32  = 64,000,000 B
  __bf16* SPK2 = (__bf16*)(ws + 297653248);      //  1600*10000 bf16 = 32,000,000 B
  float*  CUR3 = (float*) (ws + 329653248);      //  1600*5 f32      = 32,000 B

  snn_layer1 <<<(BSZ * LDK) / 256, 256, 0, stream>>>(x, SPK1);
  snn_padzero<<<((MPAD - MROWS) * LDK / 8) / 256, 256, 0, stream>>>(SPK1);
  snn_w1bf16 <<<((size_t)NH * (LDK / 4) + 255) / 256, 256, 0, stream>>>(W1, W1b);
  snn_gemm1  <<<NH / 16, 256, 0, stream>>>(SPK1, W1b, CUR2);
  snn_layer2 <<<(BSZ * NH) / 256, 256, 0, stream>>>(CUR2, SPK2);
  snn_cur3   <<<MROWS, 256, 0, stream>>>(SPK2, W2, CUR3);
  snn_layer3 <<<1, 128, 0, stream>>>(CUR3, out);
}
